// ConsistencyModel_72722386256242
// MI455X (gfx1250) — matrix operand order repaired and checked by reading, not compiled, not run on hardware
//
#include <hip/hip_runtime.h>
#include <hip/hip_bf16.h>
#include <math.h>

typedef _Float16 h16;
typedef __attribute__((ext_vector_type(8)))  _Float16 h8v;
typedef __attribute__((ext_vector_type(16))) _Float16 v16h;
typedef __attribute__((ext_vector_type(8)))  float    v8f;

#define B_ROWS 32768
#define STATE  256
#define ACT    64
#define TD     16
#define NE     16
#define H      256
#define IN_DIM 336
#define IN_PAD 384
#define BM     64      // rows per block tile

union ABFrag { v16h v; h8v h[2]; };

// A fragment (16x32, f16) from row-major [*, ld] buffer.
__device__ __forceinline__ v16h frag_a(const h16* base, int ld, int row0, int k0, int lane) {
  int ln = lane & 15, hg = lane >> 4;
  const h16* p = base + (size_t)(row0 + ln) * ld + (k0 + 8 * hg);
  ABFrag f;
  f.h[0] = *(const h8v*)p;
  f.h[1] = *(const h8v*)(p + 16);
  return f.v;
}

// B fragment (32x16, f16) from output-major weights Wt[o][k].
__device__ __forceinline__ v16h frag_b(const h16* base, int ldk, int col0, int k0, int lane) {
  int ln = lane & 15, hg = lane >> 4;
  const h16* p = base + (size_t)(col0 + ln) * ldk + (k0 + 8 * hg);
  ABFrag f;
  f.h[0] = *(const h8v*)p;
  f.h[1] = *(const h8v*)(p + 16);
  return f.v;
}

__device__ __forceinline__ v8f wmma16(v16h a, v16h b, v8f c) {
  return __builtin_amdgcn_wmma_f32_16x16x32_f16(false, a, false, b, (short)0, c, false, false);
}

// 2x4 register-blocked GEMM: one wave computes a 32-row strip against 4 n-tiles
// located at n0+{0,16,128,144}. 8 WMMA per (2 A-frag + 4 B-frag) loads.
__device__ __forceinline__ void gemm_2x4(const h16* A, int lda, int m0,
                                         const h16* W, int ldk, int n0,
                                         int K, int lane, v8f acc[8]) {
  for (int k0 = 0; k0 < K; k0 += 32) {
    v16h a0 = frag_a(A, lda, m0,      k0, lane);
    v16h a1 = frag_a(A, lda, m0 + 16, k0, lane);
    v16h b0 = frag_b(W, ldk, n0,       k0, lane);
    v16h b1 = frag_b(W, ldk, n0 + 16,  k0, lane);
    v16h b2 = frag_b(W, ldk, n0 + 128, k0, lane);
    v16h b3 = frag_b(W, ldk, n0 + 144, k0, lane);
    acc[0] = wmma16(a0, b0, acc[0]);
    acc[1] = wmma16(a0, b1, acc[1]);
    acc[2] = wmma16(a0, b2, acc[2]);
    acc[3] = wmma16(a0, b3, acc[3]);
    acc[4] = wmma16(a1, b0, acc[4]);
    acc[5] = wmma16(a1, b1, acc[5]);
    acc[6] = wmma16(a1, b2, acc[6]);
    acc[7] = wmma16(a1, b3, acc[7]);
  }
}

// Fast exact mish: tanh(softplus(x)) = (t^2+2t)/(t^2+2t+2), t=e^x.
// Written as 1 - 2/(t*(t+2)+2): overflow-safe (t->inf => 1), t->0 => 0.
__device__ __forceinline__ float mishf(float x) {
  float t = __expf(x);
  float d = __fmaf_rn(t, t + 2.0f, 2.0f);
  float r = __fmaf_rn(-2.0f, __frcp_rn(d), 1.0f);
  return x * r;
}

// ---- weight convert: src [batch][ksrc][ocols] f32 -> dst [batch][ocols][kpad] f16 (zero pad K)
__global__ void k_wt(const float* __restrict__ src, h16* __restrict__ dst,
                     int ksrc, int ocols, int kpad, long total) {
  long i = (long)blockIdx.x * blockDim.x + threadIdx.x;
  if (i >= total) return;
  int  k = (int)(i % kpad);
  long r = i / kpad;
  int  o = (int)(r % ocols);
  long b = r / ocols;
  float v = (k < ksrc) ? src[(b * ksrc + k) * ocols + o] : 0.0f;
  dst[i] = (h16)v;
}

__global__ void k_zero(float* __restrict__ p, long n) {
  long i = (long)blockIdx.x * blockDim.x + threadIdx.x;
  if (i < n) p[i] = 0.0f;
}

// ---- per-row scalings + sinusoidal embedding + time MLP (16->32->16)
__global__ void k_prep(const float* __restrict__ sigma,
                       const float* __restrict__ tW1, const float* __restrict__ tb1,
                       const float* __restrict__ tW2, const float* __restrict__ tb2,
                       float* __restrict__ tbuf, float* __restrict__ sc) {
  int b = blockIdx.x * blockDim.x + threadIdx.x;
  if (b >= B_ROWS) return;
  float s   = sigma[b];
  float sd2 = 0.25f;
  float dm  = s - 0.002f;
  float inv = rsqrtf(s * s + sd2);
  sc[b * 3 + 0] = sd2 / (dm * dm + sd2);   // c_skip
  sc[b * 3 + 1] = dm * 0.5f * inv;         // c_out
  sc[b * 3 + 2] = inv;                     // c_in
  float rt = 250.0f * __logf(s + 1e-44f);

  const float NEGL = -9.210340371976184f / 7.0f;
  float pe[16];
#pragma unroll
  for (int j = 0; j < 8; j++) {
    float a = rt * __expf(NEGL * (float)j);
    pe[j] = __sinf(a);
    pe[j + 8] = __cosf(a);
  }
  float hm[32];
  for (int j = 0; j < 32; j++) {
    float acc = tb1[j];
#pragma unroll
    for (int i = 0; i < 16; i++) acc += pe[i] * tW1[i * 32 + j];
    hm[j] = mishf(acc);
  }
  for (int k = 0; k < 16; k++) {
    float acc = tb2[k];
#pragma unroll
    for (int j = 0; j < 32; j++) acc += hm[j] * tW2[j * 16 + k];
    tbuf[b * 16 + k] = acc;
  }
}

// ---- build padded f16 input [B][384] = [c_in*x | t | state | 0]
__global__ void k_build(const float* __restrict__ x, const float* __restrict__ state,
                        const float* __restrict__ tbuf, const float* __restrict__ sc,
                        h16* __restrict__ inp) {
  long i = (long)blockIdx.x * blockDim.x + threadIdx.x;
  if (i >= (long)B_ROWS * IN_PAD) return;
  int c = (int)(i % IN_PAD);
  int b = (int)(i / IN_PAD);
  float v;
  if (c < ACT)              v = sc[b * 3 + 2] * x[(size_t)b * ACT + c];
  else if (c < ACT + TD)    v = tbuf[b * TD + (c - ACT)];
  else if (c < IN_DIM)      v = state[(size_t)b * STATE + (c - ACT - TD)];
  else                      v = 0.0f;
  inp[i] = (h16)v;
}

// ---- gate: WMMA layer1 (384->256, relu), WMMA layer2 (256->16), softmax + top4
__global__ void __launch_bounds__(256)
k_gate(const h16* __restrict__ inp, const h16* __restrict__ gw1t, const float* __restrict__ gb1,
       const h16* __restrict__ gw2t, const float* __restrict__ gb2,
       float* __restrict__ wgt, float* __restrict__ acc) {
  extern __shared__ char smem[];
  h16*   sA  = (h16*)smem;                 // [64][384]
  h16*   sG1 = sA + BM * IN_PAD;           // [64][256]
  float* sG  = (float*)(sG1 + BM * H);     // [64][16]
  int tid = threadIdx.x, lane = tid & 31, wid = tid >> 5;
  int ln = lane & 15, hg = lane >> 4;
  int row0 = blockIdx.x * BM;
  const int noff[4] = {0, 16, 128, 144};

  {
    const h8v* g = (const h8v*)(inp + (size_t)row0 * IN_PAD);
    h8v* d = (h8v*)sA;
    for (int i = tid; i < BM * IN_PAD / 8; i += 256) d[i] = g[i];
  }
  __syncthreads();

  // layer 1: [64][256], one 2x4 block per wave
  {
    int m0 = (wid & 1) * 32, n0 = (wid >> 1) * 32;
    v8f z = {};
    v8f a8[8];
#pragma unroll
    for (int i = 0; i < 8; i++) a8[i] = z;
    gemm_2x4(sA, IN_PAD, m0, gw1t, IN_PAD, n0, IN_PAD, lane, a8);
#pragma unroll
    for (int mi = 0; mi < 2; mi++)
#pragma unroll
      for (int ni = 0; ni < 4; ni++) {
        v8f c = a8[mi * 4 + ni];
        int col = n0 + noff[ni] + ln;
        float bias = gb1[col];
#pragma unroll
        for (int r = 0; r < 8; r++) {
          float v = c[r] + bias;
          sG1[(m0 + mi * 16 + r + 8 * hg) * H + col] = (h16)(v > 0.0f ? v : 0.0f);
        }
      }
  }
  __syncthreads();

  // layer 2: [64][16]
  if (wid < 4) {
    int m0 = wid << 4;
    v8f c = {};
    for (int k0 = 0; k0 < H; k0 += 32)
      c = wmma16(frag_a(sG1, H, m0, k0, lane), frag_b(gw2t, H, 0, k0, lane), c);
    float bias = gb2[ln];
#pragma unroll
    for (int r = 0; r < 8; r++) sG[(m0 + r + 8 * hg) * 16 + ln] = c[r] + bias;
  }
  __syncthreads();

  // softmax + top4 per row
  if (tid < BM) {
    int b = row0 + tid;
    float p[16], mx = -1e30f;
#pragma unroll
    for (int e = 0; e < 16; e++) { p[e] = sG[tid * 16 + e]; mx = fmaxf(mx, p[e]); }
    float sum = 0.0f;
#pragma unroll
    for (int e = 0; e < 16; e++) { p[e] = __expf(p[e] - mx); sum += p[e]; }
    float is = 1.0f / sum, ent = 0.0f;
#pragma unroll
    for (int e = 0; e < 16; e++) { p[e] *= is; ent -= p[e] * __logf(p[e] + 1e-9f); }
    float w[16], tmp[16];
#pragma unroll
    for (int e = 0; e < 16; e++) { w[e] = 0.0f; tmp[e] = p[e]; }
    int idx[4];
    float tsum = 1e-9f;
#pragma unroll
    for (int k = 0; k < 4; k++) {
      int bj = 0; float bv = -1.0f;
#pragma unroll
      for (int e = 0; e < 16; e++) if (tmp[e] > bv) { bv = tmp[e]; bj = e; }
      idx[k] = bj; w[bj] = bv; tsum += bv; tmp[bj] = -2.0f;
    }
    float itn = 1.0f / tsum;
#pragma unroll
    for (int e = 0; e < 16; e++) wgt[(size_t)b * 16 + e] = w[e] * itn;
    atomicAdd(&acc[0], ent);
#pragma unroll
    for (int k = 0; k < 4; k++) atomicAdd(&acc[1 + idx[k]], 1.0f);
  }
}

// ---- experts: 3 fused mish-GEMM layers per (row-tile, expert), weighted atomic combine
__global__ void __launch_bounds__(256)
k_expert(const h16* __restrict__ inp,
         const h16* __restrict__ w1t, const float* __restrict__ eb1,
         const h16* __restrict__ w2t, const float* __restrict__ eb2,
         const h16* __restrict__ w3t, const float* __restrict__ eb3,
         const float* __restrict__ wgt, float* __restrict__ comb) {
  extern __shared__ char smem[];
  h16* sA  = (h16*)smem;            // [64][384]
  h16* sH1 = sA + BM * IN_PAD;      // [64][256]
  h16* sH2 = sH1 + BM * H;          // [64][256]
  int tid = threadIdx.x, lane = tid & 31, wid = tid >> 5;
  int ln = lane & 15, hg = lane >> 4;
  int row0 = blockIdx.x * BM;
  int e = blockIdx.y;
  const h16* W1 = w1t + (size_t)e * H * IN_PAD;
  const h16* W2 = w2t + (size_t)e * H * H;
  const h16* W3 = w3t + (size_t)e * H * H;
  const int noff[4] = {0, 16, 128, 144};
  int m0 = (wid & 1) * 32, n0 = (wid >> 1) * 32;
  v8f z = {};

  {
    const h8v* g = (const h8v*)(inp + (size_t)row0 * IN_PAD);
    h8v* d = (h8v*)sA;
    for (int i = tid; i < BM * IN_PAD / 8; i += 256) d[i] = g[i];
  }
  __syncthreads();

  // layer 1: inp(384) -> H, mish
  {
    v8f a8[8];
#pragma unroll
    for (int i = 0; i < 8; i++) a8[i] = z;
    gemm_2x4(sA, IN_PAD, m0, W1, IN_PAD, n0, IN_PAD, lane, a8);
#pragma unroll
    for (int mi = 0; mi < 2; mi++)
#pragma unroll
      for (int ni = 0; ni < 4; ni++) {
        v8f c = a8[mi * 4 + ni];
        int col = n0 + noff[ni] + ln;
        float bias = eb1[e * H + col];
#pragma unroll
        for (int r = 0; r < 8; r++)
          sH1[(m0 + mi * 16 + r + 8 * hg) * H + col] = (h16)mishf(c[r] + bias);
      }
  }
  __syncthreads();

  // layer 2: H -> H, mish
  {
    v8f a8[8];
#pragma unroll
    for (int i = 0; i < 8; i++) a8[i] = z;
    gemm_2x4(sH1, H, m0, W2, H, n0, H, lane, a8);
#pragma unroll
    for (int mi = 0; mi < 2; mi++)
#pragma unroll
      for (int ni = 0; ni < 4; ni++) {
        v8f c = a8[mi * 4 + ni];
        int col = n0 + noff[ni] + ln;
        float bias = eb2[e * H + col];
#pragma unroll
        for (int r = 0; r < 8; r++)
          sH2[(m0 + mi * 16 + r + 8 * hg) * H + col] = (h16)mishf(c[r] + bias);
      }
  }
  __syncthreads();

  // layer 3: H -> H, mish, scale by gate weight, atomic combine
  {
    v8f a8[8];
#pragma unroll
    for (int i = 0; i < 8; i++) a8[i] = z;
    gemm_2x4(sH2, H, m0, W3, H, n0, H, lane, a8);
#pragma unroll
    for (int mi = 0; mi < 2; mi++)
#pragma unroll
      for (int ni = 0; ni < 4; ni++) {
        v8f c = a8[mi * 4 + ni];
        int col = n0 + noff[ni] + ln;
        float bias = eb3[e * H + col];
#pragma unroll
        for (int r = 0; r < 8; r++) {
          int row = row0 + m0 + mi * 16 + r + 8 * hg;
          float gw = wgt[(size_t)row * 16 + e];
          if (gw != 0.0f)
            atomicAdd(&comb[(size_t)row * H + col], mishf(c[r] + bias) * gw);
        }
      }
  }
}

// ---- final: comb @ fW + fb, consistency scalings, clip
__global__ void __launch_bounds__(256)
k_final(const float* __restrict__ comb, const h16* __restrict__ fwt, const float* __restrict__ fb,
        const float* __restrict__ sc, const float* __restrict__ x, float* __restrict__ out) {
  extern __shared__ char smem[];
  h16* sC = (h16*)smem;   // [64][256]
  int tid = threadIdx.x, lane = tid & 31, wid = tid >> 5;
  int ln = lane & 15, hg = lane >> 4;
  int row0 = blockIdx.x * BM;
  for (int i = tid; i < BM * H; i += 256) sC[i] = (h16)comb[(size_t)row0 * H + i];
  __syncthreads();

  // 4m x 4n tiles; waves 0..3 take one 2x2 block each
  for (int bi = wid; bi < 4; bi += 8) {
    int m0 = (bi & 1) * 32, n0 = (bi >> 1) * 32;
    v8f c00 = {}, c01 = {}, c10 = {}, c11 = {};
    for (int k0 = 0; k0 < H; k0 += 32) {
      v16h a0 = frag_a(sC, H, m0,      k0, lane);
      v16h a1 = frag_a(sC, H, m0 + 16, k0, lane);
      v16h b0 = frag_b(fwt, H, n0,      k0, lane);
      v16h b1 = frag_b(fwt, H, n0 + 16, k0, lane);
      c00 = wmma16(a0, b0, c00);
      c01 = wmma16(a0, b1, c01);
      c10 = wmma16(a1, b0, c10);
      c11 = wmma16(a1, b1, c11);
    }
    const v8f* cs[4] = { &c00, &c01, &c10, &c11 };
#pragma unroll
    for (int mi = 0; mi < 2; mi++)
#pragma unroll
      for (int ni = 0; ni < 2; ni++) {
        const v8f& c = *cs[mi * 2 + ni];
        int col = n0 + ni * 16 + ln;
        float bias = fb[col];
#pragma unroll
        for (int r = 0; r < 8; r++) {
          int row = row0 + m0 + mi * 16 + r + 8 * hg;
          float d = sc[row * 3 + 1] * (c[r] + bias) + sc[row * 3 + 0] * x[(size_t)row * ACT + col];
          d = fminf(fmaxf(d, -1.0f), 1.0f);
          out[(size_t)row * ACT + col] = d;
        }
      }
  }
}

// ---- aux loss: var(load, ddof=1) + mean entropy
__global__ void k_aux(const float* __restrict__ acc, float* __restrict__ out) {
  if (threadIdx.x == 0 && blockIdx.x == 0) {
    float l[16], mean = 0.0f;
#pragma unroll
    for (int e = 0; e < 16; e++) { l[e] = acc[1 + e] / ((float)B_ROWS + 1e-9f); mean += l[e]; }
    mean *= (1.0f / 16.0f);
    float var = 0.0f;
#pragma unroll
    for (int e = 0; e < 16; e++) { float d = l[e] - mean; var += d * d; }
    var *= (1.0f / 15.0f);
    out[(size_t)B_ROWS * ACT] = var + acc[0] / (float)B_ROWS;
  }
}

extern "C" void kernel_launch(void* const* d_in, const int* in_sizes, int n_in,
                              void* d_out, int out_size, void* d_ws, size_t ws_size,
                              hipStream_t stream) {
  const float* x     = (const float*)d_in[0];
  const float* sigma = (const float*)d_in[1];
  const float* state = (const float*)d_in[2];
  const float* tW1   = (const float*)d_in[3];
  const float* tb1   = (const float*)d_in[4];
  const float* tW2   = (const float*)d_in[5];
  const float* tb2   = (const float*)d_in[6];
  const float* gW1   = (const float*)d_in[7];
  const float* gb1   = (const float*)d_in[8];
  const float* gW2   = (const float*)d_in[9];
  const float* gb2   = (const float*)d_in[10];
  const float* eW1   = (const float*)d_in[11];
  const float* eb1   = (const float*)d_in[12];
  const float* eW2   = (const float*)d_in[13];
  const float* eb2   = (const float*)d_in[14];
  const float* eW3   = (const float*)d_in[15];
  const float* eb3   = (const float*)d_in[16];
  const float* fW    = (const float*)d_in[17];
  const float* fb    = (const float*)d_in[18];
  float* out = (float*)d_out;

  char* ws = (char*)d_ws;
  size_t off = 0;
  auto alloc = [&](size_t bytes) -> char* {
    char* p = ws + off;
    off += (bytes + 255) & ~(size_t)255;
    return p;
  };
  h16*   inp  = (h16*)alloc((size_t)B_ROWS * IN_PAD * 2);
  h16*   w1t  = (h16*)alloc((size_t)NE * H * IN_PAD * 2);
  h16*   w2t  = (h16*)alloc((size_t)NE * H * H * 2);
  h16*   w3t  = (h16*)alloc((size_t)NE * H * H * 2);
  h16*   gw1t = (h16*)alloc((size_t)H * IN_PAD * 2);
  h16*   gw2t = (h16*)alloc((size_t)NE * H * 2);
  h16*   fwt  = (h16*)alloc((size_t)ACT * H * 2);
  float* tbuf = (float*)alloc((size_t)B_ROWS * TD * 4);
  float* sc   = (float*)alloc((size_t)B_ROWS * 3 * 4);
  float* wgt  = (float*)alloc((size_t)B_ROWS * NE * 4);
  float* comb = (float*)alloc((size_t)B_ROWS * H * 4);
  float* acc  = (float*)alloc(32 * 4);

  // weight conversion (f32 -> f16, output-major, K padded)
  { long n = (long)NE * H * IN_PAD; k_wt<<<(n + 255) / 256, 256, 0, stream>>>(eW1, w1t, IN_DIM, H, IN_PAD, n); }
  { long n = (long)NE * H * H;      k_wt<<<(n + 255) / 256, 256, 0, stream>>>(eW2, w2t, H, H, H, n); }
  { long n = (long)NE * H * H;      k_wt<<<(n + 255) / 256, 256, 0, stream>>>(eW3, w3t, H, H, H, n); }
  { long n = (long)H * IN_PAD;      k_wt<<<(n + 255) / 256, 256, 0, stream>>>(gW1, gw1t, IN_DIM, H, IN_PAD, n); }
  { long n = (long)NE * H;          k_wt<<<(n + 255) / 256, 256, 0, stream>>>(gW2, gw2t, H, NE, H, n); }
  { long n = (long)ACT * H;         k_wt<<<(n + 255) / 256, 256, 0, stream>>>(fW, fwt, H, ACT, H, n); }

  // zero accumulation buffers
  { long n = (long)B_ROWS * H; k_zero<<<(n + 255) / 256, 256, 0, stream>>>(comb, n); }
  k_zero<<<1, 32, 0, stream>>>(acc, 32);

  // prologue
  k_prep<<<B_ROWS / 256, 256, 0, stream>>>(sigma, tW1, tb1, tW2, tb2, tbuf, sc);
  { long n = (long)B_ROWS * IN_PAD; k_build<<<(n + 255) / 256, 256, 0, stream>>>(x, state, tbuf, sc, inp); }

  // gate
  size_t smem_gate = (size_t)BM * IN_PAD * 2 + (size_t)BM * H * 2 + (size_t)BM * 16 * 4;
  k_gate<<<B_ROWS / BM, 256, smem_gate, stream>>>(inp, gw1t, gb1, gw2t, gb2, wgt, acc);

  // experts
  size_t smem_exp = (size_t)BM * IN_PAD * 2 + 2 * ((size_t)BM * H * 2);
  dim3 g3(B_ROWS / BM, NE);
  k_expert<<<g3, 256, smem_exp, stream>>>(inp, w1t, eb1, w2t, eb2, w3t, eb3, wgt, comb);

  // final + aux
  size_t smem_fin = (size_t)BM * H * 2;
  k_final<<<B_ROWS / BM, 256, smem_fin, stream>>>(comb, fwt, fb, sc, x, out);
  k_aux<<<1, 32, 0, stream>>>(acc, out);
}
